// MeshEncoder_44083544326930
// MI455X (gfx1250) — compile-verified, matrix operand order checked by reading, not run on hardware
//
#include <hip/hip_runtime.h>
#include <stdint.h>

typedef _Float16 half_t;
typedef __attribute__((ext_vector_type(16))) _Float16 v16h;
typedef __attribute__((ext_vector_type(8)))  _Float16 v8h;
typedef __attribute__((ext_vector_type(8)))  float    v8f;
typedef __attribute__((ext_vector_type(4)))  unsigned int v4u;
typedef __attribute__((ext_vector_type(8)))  int v8i;
typedef __attribute__((ext_vector_type(4)))  int v4i;

#define MESHES 8
#define NV 4096
#define NF 8192
#define WORDS (NV / 32)   // 128 u32 words per adjacency row
#define AS 512            // activation row stride (half elements)
#define SS 512            // support row stride (float elements)
#define NLAYERS 17
#define LATENT 512

#define ASPACE_G __attribute__((address_space(1)))
#define ASPACE_L __attribute__((address_space(3)))

// --- capability probes (reported via stderr warnings) ----------------------
#if __has_builtin(__builtin_amdgcn_tensor_load_to_lds)
#warning "probe: __builtin_amdgcn_tensor_load_to_lds PRESENT"
#else
#warning "probe: __builtin_amdgcn_tensor_load_to_lds ABSENT"
#endif
#if __has_builtin(__builtin_amdgcn_global_load_async_to_lds_b128)
#warning "probe: __builtin_amdgcn_global_load_async_to_lds_b128 PRESENT"
#else
#warning "probe: __builtin_amdgcn_global_load_async_to_lds_b128 ABSENT"
#endif

#if defined(__gfx1250__) && __has_builtin(__builtin_amdgcn_tensor_load_to_lds) && __has_builtin(__builtin_amdgcn_s_wait_tensorcnt)
#define HAVE_TDM 1
#else
#define HAVE_TDM 0
#endif
#if defined(__gfx1250__) && __has_builtin(__builtin_amdgcn_global_load_async_to_lds_b128) && __has_builtin(__builtin_amdgcn_s_wait_asynccnt)
#define HAVE_ASYNC 1
#else
#define HAVE_ASYNC 0
#endif

// ---------------------------------------------------------------------------
// Adjacency: bitmask (idempotent atomicOr -> deterministic), diag preset.
// ---------------------------------------------------------------------------
__global__ __launch_bounds__(256) void k_init_adj(uint32_t* __restrict__ adj) {
    int idx = blockIdx.x * 256 + threadIdx.x;              // [0, MESHES*NV*WORDS)
    if (idx >= MESHES * NV * WORDS) return;
    int w   = idx & (WORDS - 1);
    int row = (idx / WORDS) & (NV - 1);
    adj[idx] = (w == (row >> 5)) ? (1u << (row & 31)) : 0u; // self-loop bit
}

__global__ __launch_bounds__(256) void k_build_adj(const int* __restrict__ faces,
                                                   uint32_t* __restrict__ adj) {
    int idx = blockIdx.x * 256 + threadIdx.x;              // [0, MESHES*NF)
    if (idx >= MESHES * NF) return;
    int m = idx / NF;
    const int* f = faces + (size_t)idx * 3;
    int a = f[0], b = f[1], c = f[2];
    uint32_t* base = adj + (size_t)m * NV * WORDS;
    atomicOr(&base[(size_t)a * WORDS + (b >> 5)], 1u << (b & 31));
    atomicOr(&base[(size_t)a * WORDS + (c >> 5)], 1u << (c & 31));
    atomicOr(&base[(size_t)b * WORDS + (a >> 5)], 1u << (a & 31));
    atomicOr(&base[(size_t)b * WORDS + (c >> 5)], 1u << (c & 31));
    atomicOr(&base[(size_t)c * WORDS + (a >> 5)], 1u << (a & 31));
    atomicOr(&base[(size_t)c * WORDS + (b >> 5)], 1u << (b & 31));
}

__global__ __launch_bounds__(256) void k_invdeg(const uint32_t* __restrict__ adj,
                                                float* __restrict__ invdeg) {
    int idx = blockIdx.x * 256 + threadIdx.x;              // [0, MESHES*NV)
    if (idx >= MESHES * NV) return;
    const uint32_t* bits = adj + (size_t)idx * WORDS;
    int cnt = 0;
    for (int w = 0; w < WORDS; ++w) cnt += __popc(bits[w]);
    invdeg[idx] = 1.0f / (float)cnt;                       // >=1 (self-loop)
}

// ---------------------------------------------------------------------------
// Weight conversion: W[fin][fout] f32 -> Wt[foutp16][finp32] f16, transposed,
// zero padded so WMMA B-fragments are contiguous 16B loads per lane.
// ---------------------------------------------------------------------------
__global__ __launch_bounds__(256) void k_conv_w(const float* __restrict__ W,
                                                half_t* __restrict__ Wt,
                                                int fin, int fout, int finp, int total) {
    int idx = blockIdx.x * 256 + threadIdx.x;
    if (idx >= total) return;
    int r = idx / finp, c = idx - r * finp;                // r: out col, c: k
    Wt[idx] = (r < fout && c < fin) ? (half_t)W[(size_t)c * fout + r] : (half_t)0.f;
}

// verts [M][V][3] f32 -> actA f16 rows, cols 3..31 zeroed (K pad for layer 0)
__global__ __launch_bounds__(256) void k_init_act(const float* __restrict__ verts,
                                                  half_t* __restrict__ act) {
    int idx = blockIdx.x * 256 + threadIdx.x;              // [0, MESHES*NV*32)
    if (idx >= MESHES * NV * 32) return;
    int c = idx & 31;
    int rv = idx >> 5;                                     // mesh*NV + row
    float v = (c < 3) ? verts[(size_t)rv * 3 + c] : 0.f;
    act[(size_t)rv * AS + c] = (half_t)v;
}

// ---------------------------------------------------------------------------
// WMMA GEMM: S[M][V][SS] (f32) = X[M][V][AS] (f16) @ Wt^T
// 8 waves / block; block = 128 rows x 16 cols.
//  - W tile (16 x finp, contiguous): TDM DMA global->LDS (1D descriptor).
//  - A tile: 128x32 K-chunks staged global->LDS with ASYNC loads,
//    double-buffered so chunk i+1's DMA overlaps chunk i's WMMA.
// Fragment layouts per CDNA5 ISA 7.12.2 (16-bit A 16x32; D = 8 VGPRs).
// ---------------------------------------------------------------------------
__global__ __launch_bounds__(256) void k_gemm(const half_t* __restrict__ X,
                                              const half_t* __restrict__ Wt,
                                              float* __restrict__ S, int finp) {
    __shared__ __attribute__((aligned(16))) half_t smemW[16 * 512];   // 16 KB
#if HAVE_ASYNC
    __shared__ __attribute__((aligned(16))) half_t smemX[2][128 * 32]; // 16 KB
#endif
    int wave = threadIdx.x >> 5;
    int lane = threadIdx.x & 31;
    int h = lane >> 4;                                     // half-wave select
    int n = lane & 15;                                     // row (A) / col (B)
    int rowTile = blockIdx.x * 8 + wave;
    int colTile = blockIdx.y;
    int mesh = blockIdx.z;
    const int N = 16 * finp;                               // W tile elements (<=8192)
    const half_t* wtile = Wt + (size_t)colTile * 16 * finp;

#if HAVE_TDM
    // --- TDM path: one wave issues a 1D tensor DMA global->LDS -------------
    if (threadIdx.x < 32) {
        uint64_t ga = (uint64_t)(uintptr_t)wtile;
        uint32_t la = (uint32_t)(uintptr_t)smemW;          // LDS aperture low bits
        v4u g0; v8i g1; v4i gz = {0, 0, 0, 0};
        // D# group0: count=1 | lds_addr | global_addr[56:0] | type=2
        g0.x = 1u;
        g0.y = la;
        g0.z = (uint32_t)ga;
        g0.w = (uint32_t)((ga >> 32) & 0x01FFFFFFu) | (2u << 30);
        // D# group1: wg_mask=0, data_size=1 (2B), tensor_dim0 = tile_dim0 = N,
        // tensor_dim1 = tile_dim1 = 1, tensor_dim0_stride = N
        g1[0] = (int)(1u << 16);                            // data_size @ [17:16]
        g1[1] = (int)(((uint32_t)N & 0xFFFFu) << 16);       // tensor_dim0 lo16 @ [63:48]
        g1[2] = (int)(((uint32_t)N >> 16) | (1u << 16));    // t_dim0 hi | tensor_dim1 lo
        g1[3] = (int)((uint32_t)N << 16);                   // tile_dim0 @ [127:112]
        g1[4] = 1;                                          // tile_dim1 = 1
        g1[5] = (int)N;                                     // tensor_dim0_stride lo32
        g1[6] = 0;
        g1[7] = 0;
#if __clang_major__ >= 23
        v8i gz8 = {0, 0, 0, 0, 0, 0, 0, 0};
        __builtin_amdgcn_tensor_load_to_lds(g0, g1, gz, gz, gz8, 0);
#else
        __builtin_amdgcn_tensor_load_to_lds(g0, g1, gz, gz, 0);
#endif
        __builtin_amdgcn_s_wait_tensorcnt(0);
    }
    __syncthreads();
#else
    for (int i = threadIdx.x * 8; i < N; i += 256 * 8)
        *(v8h*)&smemW[i] = *(const v8h*)&wtile[i];
    __syncthreads();
#endif

    const half_t* WrowL = smemW + (size_t)n * finp;
    v8f acc = {};

#if HAVE_ASYNC
    // --- async double-buffered A staging: chunks of 128 rows x 32 halves ---
    const half_t* Xbase = X + ((size_t)mesh * NV + (size_t)blockIdx.x * 128) * AS;
    const int nch = finp >> 5;
    auto issue = [&](int buf, int kc) {
        // 128*32 halves = 512 x 16B pieces; 2 pieces per thread
        for (int p = threadIdx.x; p < 512; p += 256) {
            int r = p >> 2, sub = p & 3;
            const half_t* g = Xbase + (size_t)r * AS + kc * 32 + sub * 8;
            __builtin_amdgcn_global_load_async_to_lds_b128(
                (ASPACE_G v4i*)(uintptr_t)g,
                (ASPACE_L v4i*)(uint32_t)(uintptr_t)&smemX[buf][r * 32 + sub * 8],
                0, 0);
        }
    };
    issue(0, 0);
    for (int i = 0; i < nch; ++i) {
        int cur = i & 1;
        __builtin_amdgcn_s_wait_asynccnt(0);               // chunk i landed (this wave)
        __syncthreads();                                   // ...and all waves' pieces
        if (i + 1 < nch) issue(1 - cur, i + 1);            // overlap DMA with WMMA
        const half_t* Ar = &smemX[cur][(wave * 16 + n) * 32];
        v8h a0 = *(const v8h*)(Ar + h * 8);
        v8h a1 = *(const v8h*)(Ar + 16 + h * 8);
        v8h b0 = *(const v8h*)(WrowL + i * 32 + h * 8);
        v8h b1 = *(const v8h*)(WrowL + i * 32 + 16 + h * 8);
        v16h a = __builtin_shufflevector(a0, a1, 0,1,2,3,4,5,6,7,8,9,10,11,12,13,14,15);
        v16h b = __builtin_shufflevector(b0, b1, 0,1,2,3,4,5,6,7,8,9,10,11,12,13,14,15);
        acc = __builtin_amdgcn_wmma_f32_16x16x32_f16(false, a, false, b,
                                                     (short)0, acc, false, false);
        __syncthreads();                                   // buf reuse safety
    }
#else
    // --- direct global A loads (proven fallback) ---------------------------
    const half_t* Xrow = X + ((size_t)mesh * NV + (size_t)rowTile * 16 + n) * AS;
    for (int k0 = 0; k0 < finp; k0 += 32) {
        if (k0 + 32 < finp)
            __builtin_prefetch(Xrow + k0 + 32, 0, 0);      // global_prefetch_b8
        v8h a0 = *(const v8h*)(Xrow + k0 + h * 8);
        v8h a1 = *(const v8h*)(Xrow + k0 + 16 + h * 8);
        v8h b0 = *(const v8h*)(WrowL + k0 + h * 8);
        v8h b1 = *(const v8h*)(WrowL + k0 + 16 + h * 8);
        v16h a = __builtin_shufflevector(a0, a1, 0,1,2,3,4,5,6,7,8,9,10,11,12,13,14,15);
        v16h b = __builtin_shufflevector(b0, b1, 0,1,2,3,4,5,6,7,8,9,10,11,12,13,14,15);
        acc = __builtin_amdgcn_wmma_f32_16x16x32_f16(false, a, false, b,
                                                     (short)0, acc, false, false);
    }
#endif

    // D layout: VGPR r -> row 8*h + r, col = n
    float* out = S + ((size_t)mesh * NV + (size_t)rowTile * 16 + 8 * h) * SS
                   + (size_t)colTile * 16 + n;
    #pragma unroll
    for (int r = 0; r < 8; ++r) out[(size_t)r * SS] = acc[r];
}

// ---------------------------------------------------------------------------
// Ordered neighbor-list extraction from bitmask row (deterministic wave scan).
// ---------------------------------------------------------------------------
__device__ inline int build_nbrs(const uint32_t* __restrict__ bits,
                                 uint16_t* nbr, int* ncnt_s) {
    int lane = threadIdx.x & 31;
    if (threadIdx.x < 32) {
        uint32_t w[4]; int cnt = 0;
        #pragma unroll
        for (int t = 0; t < 4; ++t) { w[t] = bits[lane * 4 + t]; cnt += __popc(w[t]); }
        int scan = cnt;                                    // inclusive prefix
        #pragma unroll
        for (int d = 1; d < 32; d <<= 1) {
            int o = __shfl_up(scan, d);
            if (lane >= d) scan += o;
        }
        int p = scan - cnt;                                // exclusive offset
        #pragma unroll
        for (int t = 0; t < 4; ++t) {
            uint32_t wv = w[t];
            int base = (lane * 4 + t) * 32;
            while (wv) { int b = __ffs(wv) - 1; wv &= wv - 1; nbr[p++] = (uint16_t)(base + b); }
        }
        if (lane == 31) *ncnt_s = scan;
    }
    __syncthreads();
    return *ncnt_s;
}

// ZERON tail: side1 = adj @ (S[:, :side] * invdeg_row), concat S[:, side:],
// + bias, ELU, write f16; zero-pad cols [fout, foutp32) for next layer's K.
__global__ __launch_bounds__(256) void k_spmm_elu(const float* __restrict__ S,
                                                  const uint32_t* __restrict__ adj,
                                                  const float* __restrict__ invdeg,
                                                  const float* __restrict__ bias,
                                                  half_t* __restrict__ Y,
                                                  int fout, int side, int foutp32) {
    __shared__ uint16_t nbr[NV];
    __shared__ int ncnt;
    int row = blockIdx.x, mesh = blockIdx.y;
    int n = build_nbrs(adj + ((size_t)mesh * NV + row) * WORDS, nbr, &ncnt);
    const float* Sm = S + (size_t)mesh * NV * SS;
    const float* iv = invdeg + (size_t)mesh * NV;
    half_t* Yr = Y + ((size_t)mesh * NV + row) * AS;
    for (int c = threadIdx.x; c < foutp32; c += 256) {
        if (c >= fout) { Yr[c] = (half_t)0.f; continue; }
        float y;
        if (c < side) {
            float acc = 0.f;
            for (int k = 0; k < n; ++k) {
                int j = nbr[k];
                acc += Sm[(size_t)j * SS + c] * iv[j];
            }
            y = acc;
        } else {
            y = Sm[(size_t)row * SS + c];
        }
        y += bias[c];
        y = (y > 0.f) ? y : (expf(y) - 1.f);               // ELU
        Yr[c] = (half_t)y;
    }
}

// Reduce layer: per-row adj @ P, then order-independent atomic float max
// per column (deterministic result).
__global__ __launch_bounds__(256) void k_neg_init(float* __restrict__ p, int ntot) {
    int idx = blockIdx.x * 256 + threadIdx.x;
    if (idx < ntot) p[idx] = -3.4e38f;
}

__global__ __launch_bounds__(256) void k_spmm_max(const float* __restrict__ S,
                                                  const uint32_t* __restrict__ adj,
                                                  float* __restrict__ outmax) {
    __shared__ uint16_t nbr[NV];
    __shared__ int ncnt;
    int row = blockIdx.x, mesh = blockIdx.y;
    int n = build_nbrs(adj + ((size_t)mesh * NV + row) * WORDS, nbr, &ncnt);
    const float* Sm = S + (size_t)mesh * NV * SS;
    float* om = outmax + (size_t)mesh * LATENT;
    for (int c = threadIdx.x; c < LATENT; c += 256) {
        float acc = 0.f;
        for (int k = 0; k < n; ++k) acc += Sm[(size_t)nbr[k] * SS + c];
        // atomic float-max via monotonic integer punning
        if (acc >= 0.f) atomicMax((int*)&om[c], __float_as_int(acc));
        else            atomicMin((unsigned int*)&om[c], __float_as_uint(acc));
    }
}

__global__ __launch_bounds__(256) void k_finalize(const float* __restrict__ outmax,
                                                  const float* __restrict__ br,
                                                  float* __restrict__ out) {
    int idx = blockIdx.x * 256 + threadIdx.x;              // [0, MESHES*LATENT)
    if (idx >= MESHES * LATENT) return;
    int c = idx & (LATENT - 1);
    float m = outmax[idx] + br[c];
    out[idx] = (m > 0.f) ? m : (expf(m) - 1.f);
}

// ---------------------------------------------------------------------------
// Host orchestration
// ---------------------------------------------------------------------------
static const int LDIMS[NLAYERS][2] = {
    {3, 60}, {60, 60}, {60, 60}, {60, 60}, {60, 120}, {120, 120}, {120, 120},
    {120, 150}, {150, 200}, {200, 210}, {210, 250}, {250, 300}, {300, 300},
    {300, 300}, {300, 300}, {300, 300}, {300, 512}
};

extern "C" void kernel_launch(void* const* d_in, const int* in_sizes, int n_in,
                              void* d_out, int out_size, void* d_ws, size_t ws_size,
                              hipStream_t stream) {
    const float* verts = (const float*)d_in[0];
    const int*   faces = (const int*)d_in[1];

    uint8_t* ws = (uint8_t*)d_ws;
    size_t off = 0;
    auto alloc = [&](size_t bytes) -> uint8_t* {
        uint8_t* p = ws + off;
        off += (bytes + 255) & ~(size_t)255;
        return p;
    };
    uint32_t* adj    = (uint32_t*)alloc((size_t)MESHES * NV * WORDS * 4);   // 16 MB
    float*    invdeg = (float*)   alloc((size_t)MESHES * NV * 4);
    half_t*   actA   = (half_t*)  alloc((size_t)MESHES * NV * AS * 2);      // 32 MB
    half_t*   actB   = (half_t*)  alloc((size_t)MESHES * NV * AS * 2);      // 32 MB
    float*    supp   = (float*)   alloc((size_t)MESHES * NV * SS * 4);      // 64 MB
    float*    outmax = (float*)   alloc((size_t)MESHES * LATENT * 4);
    half_t*   Wt[NLAYERS];
    for (int l = 0; l < NLAYERS; ++l) {
        int fin = LDIMS[l][0], fout = LDIMS[l][1];
        int finp = (fin + 31) & ~31, foutp = (fout + 15) & ~15;
        Wt[l] = (half_t*)alloc((size_t)foutp * finp * 2);
    }

    { int t = MESHES * NV * WORDS; k_init_adj <<<(t + 255) / 256, 256, 0, stream>>>(adj); }
    { int t = MESHES * NF;         k_build_adj<<<(t + 255) / 256, 256, 0, stream>>>(faces, adj); }
    { int t = MESHES * NV;         k_invdeg   <<<(t + 255) / 256, 256, 0, stream>>>(adj, invdeg); }
    for (int l = 0; l < NLAYERS; ++l) {
        int fin = LDIMS[l][0], fout = LDIMS[l][1];
        int finp = (fin + 31) & ~31, foutp = (fout + 15) & ~15;
        int t = foutp * finp;
        k_conv_w<<<(t + 255) / 256, 256, 0, stream>>>((const float*)d_in[2 + 2 * l],
                                                      Wt[l], fin, fout, finp, t);
    }
    { int t = MESHES * NV * 32; k_init_act<<<(t + 255) / 256, 256, 0, stream>>>(verts, actA); }

    half_t* X = actA;
    half_t* Y = actB;
    for (int l = 0; l < NLAYERS - 1; ++l) {
        int fin = LDIMS[l][0], fout = LDIMS[l][1];
        int finp = (fin + 31) & ~31;
        int foutp16 = (fout + 15) & ~15, foutp32 = (fout + 31) & ~31;
        dim3 g(NV / 128, foutp16 / 16, MESHES);
        k_gemm<<<g, 256, 0, stream>>>(X, Wt[l], supp, finp);
        int side = fout / 3; if (side < 2) side = 2;
        dim3 g2(NV, MESHES);
        k_spmm_elu<<<g2, 256, 0, stream>>>(supp, adj, invdeg,
                                           (const float*)d_in[3 + 2 * l],
                                           Y, fout, side, foutp32);
        half_t* t = X; X = Y; Y = t;
    }
    // GCNMax reduce: P = X @ Wr ; out = elu(max_rows(adj @ P) + br)
    {
        int finp = (300 + 31) & ~31;                       // 320
        dim3 g(NV / 128, LATENT / 16, MESHES);
        k_gemm<<<g, 256, 0, stream>>>(X, Wt[16], supp, finp);
        int t = MESHES * LATENT;
        k_neg_init<<<(t + 255) / 256, 256, 0, stream>>>(outmax, t);
        dim3 g2(NV, MESHES);
        k_spmm_max<<<g2, 256, 0, stream>>>(supp, adj, outmax);
        k_finalize<<<(t + 255) / 256, 256, 0, stream>>>(outmax,
                                                        (const float*)d_in[2 + 2 * 16 + 1],
                                                        (float*)d_out);
    }
    (void)in_sizes; (void)n_in; (void)out_size; (void)ws_size;
}
